// Attention_43181601194500
// MI455X (gfx1250) — matrix operand order repaired and checked by reading, not compiled, not run on hardware
//
#include <hip/hip_runtime.h>

// ---------------------------------------------------------------------------
// MI455X (gfx1250) attention block: LN -> QKV GEMM -> flash attention -> out GEMM
// All matmuls on v_wmma_f32_16x16x32_bf16 (wave32, 16x16 tiles, K=32).
// GEMMs: double-buffered LDS software pipeline (1 barrier / K-step).
// ---------------------------------------------------------------------------

typedef __bf16 bf16;
typedef __attribute__((ext_vector_type(16))) __bf16 v16bf;
typedef __attribute__((ext_vector_type(8)))  __bf16 v8bf;
typedef __attribute__((ext_vector_type(8)))  float   v8f;

#define ATTN_SCALE 0.125f   // 64^-0.5

__device__ __forceinline__ v8f vzero() {
    v8f z = {0.f, 0.f, 0.f, 0.f, 0.f, 0.f, 0.f, 0.f};
    return z;
}

__device__ __forceinline__ v8f wmma_bf16f32(v16bf a, v16bf b, v8f c) {
    // D = A(16x32 bf16) x B(32x16 bf16) + C(16x16 f32)
    return __builtin_amdgcn_wmma_f32_16x16x32_bf16(
        false, a, false, b, (short)0, c, false, false);
}

// A-fragment (16x32, MxK): lane = row (lane&15); halves of wave hold
// K = {half*8..half*8+7, 16+half*8..16+half*8+7}  (ISA 7.12.2, 16-bit A)
__device__ __forceinline__ v16bf frag_a(const bf16* base, int stride, int lane) {
    const int half = lane >> 4;
    const bf16* p = base + (size_t)(lane & 15) * stride + half * 8;
    union { v16bf v; v8bf h[2]; } u;
    u.h[0] = *(const v8bf*)p;
    u.h[1] = *(const v8bf*)(p + 16);
    return u.v;
}

// B-fragment (32x16, KxN) sourced from an N-major (transposed) buffer:
// lane = column n (lane&15); halves hold K = {half*8..half*8+7, 16+half*8..16+half*8+7}
__device__ __forceinline__ v16bf frag_b(const bf16* base, int stride, int lane) {
    const bf16* p = base + (size_t)(lane & 15) * stride + (lane >> 4) * 8;
    union { v16bf v; v8bf h[2]; } u;
    u.h[0] = *(const v8bf*)p;
    u.h[1] = *(const v8bf*)(p + 16);
    return u.v;
}

// ---------------------------------------------------------------------------
// Kernel 0: fp32 -> bf16 transposed weight convert. out[c][r] = in[r][c].
// ---------------------------------------------------------------------------
__global__ void tconv_kernel(const float* __restrict__ in, bf16* __restrict__ out,
                             int rows, int cols) {
    int i = blockIdx.x * 256 + threadIdx.x;
    if (i >= rows * cols) return;
    int c = i / rows, r = i - c * rows;
    out[i] = (bf16)in[(size_t)r * cols + c];
}

// ---------------------------------------------------------------------------
// Kernel 1: LayerNorm of one 512-element row per block (256 threads), bf16 out.
// ---------------------------------------------------------------------------
__global__ void __launch_bounds__(256) ln_kernel(const float* __restrict__ x,
                                                 const float* __restrict__ g,
                                                 const float* __restrict__ b,
                                                 bf16* __restrict__ xn) {
    __shared__ float rs[256];
    __shared__ float rq[256];
    const int row = blockIdx.x, t = threadIdx.x;
    const float2 v = ((const float2*)(x + (size_t)row * 512))[t];
    rs[t] = v.x + v.y;
    rq[t] = v.x * v.x + v.y * v.y;
    __syncthreads();
    for (int s = 128; s > 0; s >>= 1) {
        if (t < s) { rs[t] += rs[t + s]; rq[t] += rq[t + s]; }
        __syncthreads();
    }
    const float mu  = rs[0] * (1.0f / 512.0f);
    const float var = rq[0] * (1.0f / 512.0f) - mu * mu;
    const float inv = rsqrtf(var + 1e-5f);
    bf16* o = xn + (size_t)row * 512;
    o[2 * t]     = (bf16)((v.x - mu) * inv * g[2 * t]     + b[2 * t]);
    o[2 * t + 1] = (bf16)((v.y - mu) * inv * g[2 * t + 1] + b[2 * t + 1]);
}

// ---------------------------------------------------------------------------
// Shared 128x128 GEMM core: C = A[M x 512] * Bt[N x 512]^T (both bf16).
// 256 threads = 8 waves (2 M x 4 N); each wave: 64x32 = 4x2 WMMA tiles.
// Double-buffered LDS (40 KB): global loads for tile k+1 issued before the
// WMMAs of tile k, LDS stores after -> one barrier per 32-wide K step.
// ---------------------------------------------------------------------------
__device__ __forceinline__ void gemm_core_128x128(const bf16* __restrict__ A,
                                                  const bf16* __restrict__ Bt,
                                                  int bm, int bn, v8f acc[4][2]) {
    __shared__ bf16 sA[2][128][40];
    __shared__ bf16 sB[2][128][40];
    const int t = threadIdx.x;
    const int lane = t & 31, w = t >> 5;
    const int m0 = (w & 1) * 64, n0 = (w >> 1) * 32;
    const int r0 = t >> 2, c0 = (t & 3) * 8;   // each thread: rows r0, r0+64
    #pragma unroll
    for (int ti = 0; ti < 4; ++ti)
        #pragma unroll
        for (int tj = 0; tj < 2; ++tj) acc[ti][tj] = vzero();

    v8bf ra0, ra1, rb0, rb1;
    auto stage_load = [&](int k0) {
        ra0 = *(const v8bf*)(A  + (size_t)(bm + r0)      * 512 + k0 + c0);
        ra1 = *(const v8bf*)(A  + (size_t)(bm + r0 + 64) * 512 + k0 + c0);
        rb0 = *(const v8bf*)(Bt + (size_t)(bn + r0)      * 512 + k0 + c0);
        rb1 = *(const v8bf*)(Bt + (size_t)(bn + r0 + 64) * 512 + k0 + c0);
    };
    auto stage_store = [&](int buf) {
        *(v8bf*)&sA[buf][r0][c0]      = ra0;
        *(v8bf*)&sA[buf][r0 + 64][c0] = ra1;
        *(v8bf*)&sB[buf][r0][c0]      = rb0;
        *(v8bf*)&sB[buf][r0 + 64][c0] = rb1;
    };

    stage_load(0);
    stage_store(0);
    for (int k = 0; k < 16; ++k) {
        const int buf = k & 1;
        __syncthreads();
        if (k < 15) stage_load((k + 1) * 32);   // in flight during WMMAs
        v16bf af[4], bfr[2];
        #pragma unroll
        for (int ti = 0; ti < 4; ++ti) af[ti]  = frag_a(&sA[buf][m0 + ti * 16][0], 40, lane);
        #pragma unroll
        for (int tj = 0; tj < 2; ++tj) bfr[tj] = frag_b(&sB[buf][n0 + tj * 16][0], 40, lane);
        #pragma unroll
        for (int ti = 0; ti < 4; ++ti)
            #pragma unroll
            for (int tj = 0; tj < 2; ++tj)
                acc[ti][tj] = wmma_bf16f32(af[ti], bfr[tj], acc[ti][tj]);
        if (k < 15) stage_store(buf ^ 1);
    }
}

// ---------------------------------------------------------------------------
// Kernel 2: QKV projection. C[8192 x 1536]; the Q/K/V column split and head
// index are uniform per (block, tj) tile -> scalar branch, no lane divergence.
// Heads unpacked to [b*8+h][2048][64] bf16; Q pre-scaled by d^-0.5.
// ---------------------------------------------------------------------------
__global__ void __launch_bounds__(256) gemm_qkv_kernel(const bf16* __restrict__ Xn,
                                                       const bf16* __restrict__ WqkvT,
                                                       bf16* __restrict__ Q,
                                                       bf16* __restrict__ K,
                                                       bf16* __restrict__ V) {
    v8f acc[4][2];
    const int bm = blockIdx.x * 128, bn = blockIdx.y * 128;
    gemm_core_128x128(Xn, WqkvT, bm, bn, acc);
    const int t = threadIdx.x, lane = t & 31, w = t >> 5;
    const int m0 = (w & 1) * 64, n0 = (w >> 1) * 32;
    const int half = lane >> 4, ln = lane & 15;
    #pragma unroll
    for (int tj = 0; tj < 2; ++tj) {
        const int tilecol = bn + n0 + tj * 16;      // uniform per wave
        const int which = tilecol >> 9;             // 0=Q 1=K 2=V (uniform)
        const int rem = tilecol & 511;
        const int h = rem >> 6;
        const int dd = (rem & 63) + ln;             // head-dim index
        bf16* const dst = (which == 0) ? Q : (which == 1) ? K : V;
        const float scl = (which == 0) ? ATTN_SCALE : 1.0f;
        #pragma unroll
        for (int ti = 0; ti < 4; ++ti)
            #pragma unroll
            for (int r = 0; r < 8; ++r) {
                const int gm = bm + m0 + ti * 16 + r + 8 * half;  // b*2048+n
                const int b = gm >> 11, n = gm & 2047;
                dst[(((size_t)(b * 8 + h) * 2048) + n) * 64 + dd] =
                    (bf16)(acc[ti][tj][r] * scl);
            }
    }
}

// ---------------------------------------------------------------------------
// Kernel 3: flash attention. One block = one (b,h) x one 128-row Q block.
// 8 waves x 16 Q rows. V-tile global loads are hoisted to the top of the
// kv-block iteration so they overlap the S = Q*K^T WMMAs and the softmax.
// ---------------------------------------------------------------------------
__global__ void __launch_bounds__(256) attn_kernel(const bf16* __restrict__ Q,
                                                   const bf16* __restrict__ K,
                                                   const bf16* __restrict__ V,
                                                   bf16* __restrict__ AO) {
    __shared__ bf16 sP[128][136];   // P 128x128 bf16 (8 waves x 16 rows), pad 8
    __shared__ bf16 sVt[64][136];   // V^T: [d][kv], pad 8
    const int t = threadIdx.x, lane = t & 31, w = t >> 5;
    const int half = lane >> 4, ln = lane & 15;
    const int blk = blockIdx.x;
    const int qb = blk & 15, bh = blk >> 4;             // 16 q-blocks, 32 (b,h)
    const bf16* Qb = Q + (((size_t)bh * 2048) + qb * 128 + w * 16) * 64;
    const bf16* Kb = K + (size_t)bh * 2048 * 64;
    const bf16* Vb = V + (size_t)bh * 2048 * 64;
    const int vn = t >> 1, vc = (t & 1) * 32;           // V staging coords

    v16bf qa[2];
    #pragma unroll
    for (int s = 0; s < 2; ++s) qa[s] = frag_a(Qb + s * 32, 64, lane);

    v8f o[4];
    float mx[8], sm[8];
    #pragma unroll
    for (int i = 0; i < 4; ++i) o[i] = vzero();
    #pragma unroll
    for (int r = 0; r < 8; ++r) { mx[r] = -3.0e30f; sm[r] = 0.f; }

    for (int jb = 0; jb < 16; ++jb) {
        // ---- issue V-tile loads early (128 rows x 64 d; thread: row vn, 32 d) --
        v8bf vreg[4];
        #pragma unroll
        for (int i = 0; i < 4; ++i)
            vreg[i] = *(const v8bf*)(Vb + ((size_t)jb * 128 + vn) * 64 + vc + i * 8);

        // ---- S = Q * K^T for a 16x128 strip (K frags straight from global) ----
        const bf16* Kblk = Kb + (size_t)jb * 128 * 64;
        v8f sacc[8];
        #pragma unroll
        for (int tj = 0; tj < 8; ++tj) {
            sacc[tj] = vzero();
            #pragma unroll
            for (int s = 0; s < 2; ++s) {
                v16bf kf = frag_b(Kblk + tj * 16 * 64 + s * 32, 64, lane);
                sacc[tj] = wmma_bf16f32(qa[s], kf, sacc[tj]);
            }
        }
        // ---- online softmax: row r+8*half, col tj*16+ln ----
        float rmax[8];
        #pragma unroll
        for (int r = 0; r < 8; ++r) {
            float m = sacc[0][r];
            #pragma unroll
            for (int tj = 1; tj < 8; ++tj) m = fmaxf(m, sacc[tj][r]);
            rmax[r] = m;
        }
        #pragma unroll
        for (int msk = 8; msk >= 1; msk >>= 1)
            #pragma unroll
            for (int r = 0; r < 8; ++r)
                rmax[r] = fmaxf(rmax[r], __shfl_xor(rmax[r], msk, 32));
        float fs[8];
        #pragma unroll
        for (int r = 0; r < 8; ++r) {
            const float mn = fmaxf(mx[r], rmax[r]);
            fs[r] = __expf(mx[r] - mn);
            mx[r] = mn;
        }
        #pragma unroll
        for (int i = 0; i < 4; ++i)
            #pragma unroll
            for (int r = 0; r < 8; ++r) o[i][r] *= fs[r];
        float rsum[8];
        #pragma unroll
        for (int r = 0; r < 8; ++r) rsum[r] = 0.f;
        #pragma unroll
        for (int tj = 0; tj < 8; ++tj)
            #pragma unroll
            for (int r = 0; r < 8; ++r) {
                const float p = __expf(sacc[tj][r] - mx[r]);
                sacc[tj][r] = p;
                rsum[r] += p;
            }
        #pragma unroll
        for (int msk = 8; msk >= 1; msk >>= 1)
            #pragma unroll
            for (int r = 0; r < 8; ++r) rsum[r] += __shfl_xor(rsum[r], msk, 32);
        #pragma unroll
        for (int r = 0; r < 8; ++r) sm[r] = sm[r] * fs[r] + rsum[r];

        // ---- P -> LDS (row-major 16x128 per wave) ----
        #pragma unroll
        for (int tj = 0; tj < 8; ++tj)
            #pragma unroll
            for (int r = 0; r < 8; ++r)
                sP[w * 16 + r + 8 * half][tj * 16 + ln] = (bf16)sacc[tj][r];

        // ---- scatter prefetched V tile transposed: sVt[d][kv] ----
        #pragma unroll
        for (int i = 0; i < 4; ++i)
            #pragma unroll
            for (int e = 0; e < 8; ++e)
                sVt[vc + i * 8 + e][vn] = vreg[i][e];
        __syncthreads();

        // ---- O += P * V ----
        #pragma unroll
        for (int s = 0; s < 4; ++s) {
            v16bf pa = frag_a(&sP[w * 16][s * 32], 136, lane);
            #pragma unroll
            for (int d = 0; d < 4; ++d) {
                v16bf vf = frag_b(&sVt[d * 16][s * 32], 136, lane);
                o[d] = wmma_bf16f32(pa, vf, o[d]);
            }
        }
        __syncthreads();  // guard sP/sVt reuse next iteration
    }

    // ---- normalize and write [b][n][h*64+d] bf16 ----
    const int b = bh >> 3, h = bh & 7;
    const int nbase = qb * 128 + w * 16;
    #pragma unroll
    for (int r = 0; r < 8; ++r) {
        const float inv = 1.0f / sm[r];
        const int n = nbase + r + 8 * half;
        bf16* dst = AO + ((size_t)(b * 2048 + n)) * 512 + h * 64 + ln;
        #pragma unroll
        for (int d = 0; d < 4; ++d)
            dst[d * 16] = (bf16)(o[d][r] * inv);
    }
}

// ---------------------------------------------------------------------------
// Kernel 4: output projection -> fp32 d_out [8192 x 512].
// ---------------------------------------------------------------------------
__global__ void __launch_bounds__(256) gemm_out_kernel(const bf16* __restrict__ AO,
                                                       const bf16* __restrict__ WoutT,
                                                       float* __restrict__ out) {
    v8f acc[4][2];
    const int bm = blockIdx.x * 128, bn = blockIdx.y * 128;
    gemm_core_128x128(AO, WoutT, bm, bn, acc);
    const int t = threadIdx.x, lane = t & 31, w = t >> 5;
    const int m0 = (w & 1) * 64, n0 = (w >> 1) * 32;
    const int half = lane >> 4, ln = lane & 15;
    #pragma unroll
    for (int ti = 0; ti < 4; ++ti)
        #pragma unroll
        for (int tj = 0; tj < 2; ++tj)
            #pragma unroll
            for (int r = 0; r < 8; ++r) {
                const int gm = bm + m0 + ti * 16 + r + 8 * half;
                const int gn = bn + n0 + tj * 16 + ln;
                out[(size_t)gm * 512 + gn] = acc[ti][tj][r];
            }
}

// ---------------------------------------------------------------------------
// Launch
// ---------------------------------------------------------------------------
extern "C" void kernel_launch(void* const* d_in, const int* in_sizes, int n_in,
                              void* d_out, int out_size, void* d_ws, size_t ws_size,
                              hipStream_t stream) {
    (void)in_sizes; (void)n_in; (void)out_size; (void)ws_size;
    const float* x     = (const float*)d_in[0];   // [4,2048,512]
    const float* gamma = (const float*)d_in[1];   // [512]
    const float* beta  = (const float*)d_in[2];   // [512]
    const float* w_qkv = (const float*)d_in[3];   // [512,1536]
    const float* w_out = (const float*)d_in[4];   // [512,512]
    float* out = (float*)d_out;

    char* ws = (char*)d_ws;
    bf16* wqkvT = (bf16*)(ws);                    // [1536][512]  1,572,864 B
    bf16* woutT = (bf16*)(ws +  1572864);         // [512][512]     524,288 B
    bf16* xn    = (bf16*)(ws +  2097152);         // [8192][512]  8,388,608 B
    bf16* Q     = (bf16*)(ws + 10485760);         // [32][2048][64]
    bf16* K     = (bf16*)(ws + 18874368);
    bf16* V     = (bf16*)(ws + 27262976);
    bf16* AO    = (bf16*)(ws + 35651584);         // [8192][512]

    tconv_kernel<<<(1536 * 512 + 255) / 256, 256, 0, stream>>>(w_qkv, wqkvT, 512, 1536);
    tconv_kernel<<<(512 * 512 + 255) / 256, 256, 0, stream>>>(w_out, woutT, 512, 512);
    ln_kernel<<<8192, 256, 0, stream>>>(x, gamma, beta, xn);
    gemm_qkv_kernel<<<dim3(64, 12), 256, 0, stream>>>(xn, wqkvT, Q, K, V);
    attn_kernel<<<512, 256, 0, stream>>>(Q, K, V, AO);
    gemm_out_kernel<<<dim3(64, 4), 256, 0, stream>>>(AO, woutT, out);
}
